// BlockLayer_1666447311268
// MI455X (gfx1250) — matrix operand order repaired and checked by reading, not compiled, not run on hardware
//
#include <hip/hip_runtime.h>
#include <hip/hip_bf16.h>
#include <math.h>

// ---------------------------------------------------------------------------
// CDNA5 (gfx1250) WMMA bf16 GEMM + full network pipeline.
// wave32; 8 waves/block; block tile 64x128; wave tile 16x64 (4 WMMAs/K-step);
// double-buffered LDS; guard-free main K loop; v_wmma_f32_16x16x32_bf16.
// ---------------------------------------------------------------------------

typedef __attribute__((ext_vector_type(16))) __bf16 v16bf;
typedef __attribute__((ext_vector_type(8)))  float  v8f;

#define BDIM_B 32
#define TDIM   96
#define NDIM   512
#define HDIM   48
#define PI_F   3.14159265358979323846f

__device__ inline unsigned int pack_bf2(float lo, float hi) {
  unsigned int a = __float_as_uint(lo), b = __float_as_uint(hi);
  a = (a + 0x7FFFu + ((a >> 16) & 1u)) >> 16;
  b = (b + 0x7FFFu + ((b >> 16) & 1u)) >> 16;
  return (a & 0xFFFFu) | (b << 16);
}

union FragBF { unsigned int u[8]; v16bf v; };

// Generic batched strided GEMM: C[z,m,n] = epi( sum_k A[z, m+rowOff, k] * W[z,k,n] )
// A element addr: A + z*aB + rr*aR + k*aC   (rr = m + rowOff; zero outside [rowLo,rowHi))
// W element addr: W + z*wB + k*wK + n*wN
// epi: (+bias[n]) (+add[z,m,n]) (relu)
__global__ __launch_bounds__(256)
void gemm_bf16_kernel(const float* __restrict__ A, long long aB, long long aR, long long aC,
                      const float* __restrict__ W, long long wB, long long wK, long long wN,
                      const float* __restrict__ bias,
                      const float* __restrict__ add, long long adB, long long adR,
                      float* __restrict__ C, long long cB, long long cR,
                      int M, int Nn, int K,
                      int rowOff, int rowLo, int rowHi, int relu)
{
  __shared__ unsigned int lA[2][64][17];    // [stage][m][k2]   packed bf16 K-pairs
  __shared__ unsigned int lB[2][128][17];   // [stage][n][k2]
  const int tid = threadIdx.x;
  const int z = blockIdx.z;
  const int rowBase = blockIdx.y * 64;
  const int colBase = blockIdx.x * 128;
  const float* Ab = A + (long long)z * aB;
  const float* Wb = W + (long long)z * wB;

  const int w = tid >> 5, lane = tid & 31;
  const int wm = w & 3, wn = w >> 2;          // 4 waves in M, 2 in N
  const bool fastWok = (wN == 1) && ((wK & 3LL) == 0) && (Nn >= 4) && ((Nn & 3) == 0);

  // ---- A loader: 64 rows x 16 k2-uints, 4 consecutive uints per thread ----
  const int a_m   = (tid * 4) >> 4;
  const int a_k2b = (tid * 4) & 15;
  const int a_row = rowBase + a_m;
  const int a_rr  = a_row + rowOff;
  const bool a_rv = (a_row < M) && (a_rr >= rowLo) && (a_rr < rowHi);
  const long long a_rrs = a_rv ? (long long)a_rr : (long long)rowLo;

  auto loadA = [&](int s, int k0blk, bool tail) {
    int kg = k0blk + 2 * a_k2b;
    if (!tail && aC == 1) {
      const float* p = Ab + a_rrs * aR + kg;
      float4 f0 = *(const float4*)p;
      float4 f1 = *(const float4*)(p + 4);
      __builtin_prefetch(p + 64, 0, 1);       // K+2 tile -> global_prefetch_b8
      if (!a_rv) { f0.x = f0.y = f0.z = f0.w = 0.f; f1.x = f1.y = f1.z = f1.w = 0.f; }
      lA[s][a_m][a_k2b + 0] = pack_bf2(f0.x, f0.y);
      lA[s][a_m][a_k2b + 1] = pack_bf2(f0.z, f0.w);
      lA[s][a_m][a_k2b + 2] = pack_bf2(f1.x, f1.y);
      lA[s][a_m][a_k2b + 3] = pack_bf2(f1.z, f1.w);
    } else {
      for (int i = 0; i < 4; ++i) {
        int k2 = a_k2b + i;
        int ka = k0blk + 2 * k2, kb = ka + 1;
        float fa = (a_rv && ka < K) ? Ab[a_rrs * aR + (long long)ka * aC] : 0.f;
        float fb = (a_rv && kb < K) ? Ab[a_rrs * aR + (long long)kb * aC] : 0.f;
        lA[s][a_m][k2] = pack_bf2(fa, fb);
      }
    }
  };

  // ---- W loader: 128 cols x 16 k2-uints; 2 groups of 4 consecutive cols ----
  auto loadB = [&](int s, int k0blk, bool tail) {
#pragma unroll
    for (int g = 0; g < 2; ++g) {
      int k2 = (tid >> 5) + 8 * g;            // 0..15
      int n0 = (tid & 31) * 4;                // 0..124
      int ka = k0blk + 2 * k2, kb = ka + 1;
      int col0 = colBase + n0;
      if (!tail && fastWok) {
        int cb = col0;
        if (cb > Nn - 4) cb = Nn - 4;         // 4-aligned clamp (Nn%4==0)
        const float* pa = Wb + (long long)ka * wK + cb;
        const float* pb = Wb + (long long)kb * wK + cb;
        float4 fa4 = *(const float4*)pa;
        float4 fb4 = *(const float4*)pb;
        float av[4] = {fa4.x, fa4.y, fa4.z, fa4.w};
        float bv[4] = {fb4.x, fb4.y, fb4.z, fb4.w};
#pragma unroll
        for (int j = 0; j < 4; ++j) {
          bool cv = (col0 + j) < Nn;          // if clamped, all 4 are OOB anyway
          lB[s][n0 + j][k2] = cv ? pack_bf2(av[j], bv[j]) : 0u;
        }
      } else {
        for (int j = 0; j < 4; ++j) {
          int col = col0 + j;
          bool nv = col < Nn;
          float fa = (nv && ka < K) ? Wb[(long long)ka * wK + (long long)col * wN] : 0.f;
          float fb = (nv && kb < K) ? Wb[(long long)kb * wK + (long long)col * wN] : 0.f;
          lB[s][n0 + j][k2] = pack_bf2(fa, fb);
        }
      }
    }
  };

  v8f acc0 = {}, acc1 = {}, acc2 = {}, acc3 = {};
  const int mrow = wm * 16 + (lane & 15);
  const int hi = (lane >= 16) ? 1 : 0;
  const int ncol = wn * 64 + (lane & 15);

  const int nkt = (K + 31) / 32;
  int cur = 0;
  loadA(0, 0, 32 > K);
  loadB(0, 0, 32 > K);

  for (int kt = 0; kt < nkt; ++kt) {
    __syncthreads();
    int nxt = cur ^ 1;
    if (kt + 1 < nkt) {
      int k0n = (kt + 1) * 32;
      loadA(nxt, k0n, (k0n + 32) > K);
      loadB(nxt, k0n, (k0n + 32) > K);
    }
    FragBF fa, fb0, fb1, fb2, fb3;
#pragma unroll
    for (int v = 0; v < 8; ++v) {             // A 16x32 bf16 layout (ISA 7.12.2)
      int k2 = ((v >= 4) ? 8 : 0) + (hi ? 4 : 0) + (v & 3);
      fa.u[v] = lA[cur][mrow][k2];
    }
#pragma unroll
    for (int v = 0; v < 8; ++v) {             // B 32x16 bf16 layout (same K order as A)
      int k2 = ((v >= 4) ? 8 : 0) + (hi ? 4 : 0) + (v & 3);
      fb0.u[v] = lB[cur][ncol][k2];
      fb1.u[v] = lB[cur][ncol + 16][k2];
      fb2.u[v] = lB[cur][ncol + 32][k2];
      fb3.u[v] = lB[cur][ncol + 48][k2];
    }
    acc0 = __builtin_amdgcn_wmma_f32_16x16x32_bf16(false, fa.v, false, fb0.v, (short)0, acc0, false, false);
    acc1 = __builtin_amdgcn_wmma_f32_16x16x32_bf16(false, fa.v, false, fb1.v, (short)0, acc1, false, false);
    acc2 = __builtin_amdgcn_wmma_f32_16x16x32_bf16(false, fa.v, false, fb2.v, (short)0, acc2, false, false);
    acc3 = __builtin_amdgcn_wmma_f32_16x16x32_bf16(false, fa.v, false, fb3.v, (short)0, acc3, false, false);
    cur = nxt;
  }

  float* Cb = C + (long long)z * cB;
  const float* addb = add ? (add + (long long)z * adB) : nullptr;
  const int lane15 = lane & 15;
  const int mofs = (lane >= 16) ? 8 : 0;
#pragma unroll
  for (int r = 0; r < 8; ++r) {               // C 16x16 f32 layout (ISA 7.12.2)
    int m = rowBase + wm * 16 + r + mofs;
    if (m >= M) continue;
    float vq[4] = {acc0[r], acc1[r], acc2[r], acc3[r]};
#pragma unroll
    for (int q = 0; q < 4; ++q) {
      int n = colBase + wn * 64 + q * 16 + lane15;
      if (n < Nn) {
        float v = vq[q];
        if (bias) v += bias[n];
        if (addb) v += addb[(long long)m * adR + n];
        if (relu) v = v > 0.f ? v : 0.f;
        Cb[(long long)m * cR + n] = v;
      }
    }
  }
}

static inline void gemm(hipStream_t st,
                        const float* A, long long aB, long long aR, long long aC,
                        const float* W, long long wB, long long wK, long long wN,
                        const float* bias,
                        const float* add, long long adB, long long adR,
                        float* C, long long cB, long long cR,
                        int M, int Nn, int K, int batches, int relu,
                        int rowOff = 0, int rowLo = 0, int rowHi = (1 << 30))
{
  dim3 grid((Nn + 127) / 128, (M + 63) / 64, batches);
  gemm_bf16_kernel<<<grid, 256, 0, st>>>(A, aB, aR, aC, W, wB, wK, wN, bias,
                                         add, adB, adR, C, cB, cR,
                                         M, Nn, K, rowOff, rowLo, rowHi, relu);
}

// ------------------------------- small kernels ------------------------------

__global__ void gen_dft_kernel(float* Cr, float* Ci) {
  int i = blockIdx.x * blockDim.x + threadIdx.x;
  if (i >= TDIM * TDIM) return;
  int t = i / TDIM, tp = i % TDIM;
  float ang = -2.f * PI_F * (float)(t * tp) / (float)TDIM;
  Cr[i] = cosf(ang);
  Ci[i] = sinf(ang);   // imag part of e^{-i theta}
}

__global__ void transpose_btn_kernel(const float* __restrict__ x, float* __restrict__ xT) {
  long long i = (long long)blockIdx.x * blockDim.x + threadIdx.x;
  if (i >= (long long)BDIM_B * TDIM * NDIM) return;
  int b = (int)(i / (TDIM * NDIM));
  int r = (int)(i % (TDIM * NDIM));
  int t = r / NDIM, n = r % NDIM;
  xT[((long long)b * NDIM + n) * TDIM + t] = x[i];
}

__global__ void ifft24_kernel(const float* __restrict__ R, const float* __restrict__ I,
                              float* __restrict__ out) {
  long long i = (long long)blockIdx.x * blockDim.x + threadIdx.x;
  const long long total = (long long)BDIM_B * 4 * NDIM * 24;
  if (i >= total) return;
  int b = (int)(i / (4 * NDIM * 24));
  int r = (int)(i % (4 * NDIM * 24));
  int k = r / (NDIM * 24);
  int r2 = r % (NDIM * 24);
  int n = r2 / 24, cp = r2 % 24;
  const float* Rp = R + ((long long)b * NDIM + n) * TDIM + k * 24;
  const float* Ip = I + ((long long)b * NDIM + n) * TDIM + k * 24;
  float s = 0.f;
  for (int c = 0; c < 24; ++c) {
    float ang = 2.f * PI_F * (float)(c * cp) / 24.f;
    s += Rp[c] * cosf(ang) - Ip[c] * sinf(ang);
  }
  s *= (1.f / 24.f);
  int inner = (k * NDIM + n) * 24 + cp;     // raw reshape (B,4,N,24) -> (B,N,96)
  int n2 = inner / TDIM, t2 = inner % TDIM;
  out[((long long)b * NDIM + n2) * TDIM + t2] = s;
}

__global__ void meanvar_kernel(const float* __restrict__ xT, float* __restrict__ xn,
                               float* __restrict__ meanb, float* __restrict__ stdb) {
  int i = blockIdx.x * blockDim.x + threadIdx.x;
  if (i >= BDIM_B * NDIM) return;
  const float* p = xT + (long long)i * TDIM;
  float s = 0.f, s2 = 0.f;
  for (int t = 0; t < TDIM; ++t) { float v = p[t]; s += v; s2 += v * v; }
  float mean = s / (float)TDIM;
  float var = s2 / (float)TDIM - mean * mean;
  float sd = sqrtf(var + 1e-5f);
  meanb[i] = mean; stdb[i] = sd;
  float inv = 1.f / sd;
  float* q = xn + (long long)i * TDIM;
  for (int t = 0; t < TDIM; ++t) q[t] = (p[t] - mean) * inv;
}

__global__ void ispec_kernel(const float* __restrict__ spec, const float* __restrict__ sw,
                             float* __restrict__ out) {
  int i = blockIdx.x * blockDim.x + threadIdx.x;
  if (i >= BDIM_B * NDIM) return;
  int n = i % NDIM;
  const float* p = spec + (long long)i * TDIM;
  float s = 0.f;
  for (int t = 0; t < TDIM; ++t) s += p[t];
  float* q = out + (long long)i * TDIM;
  for (int t = 0; t < TDIM; ++t) q[t] = s * sw[n * TDIM + t];
}

__global__ void concat_enc_kernel(const float* __restrict__ xn, const float* __restrict__ feat,
                                  float* __restrict__ h0) {
  long long i = (long long)blockIdx.x * blockDim.x + threadIdx.x;
  const long long total = (long long)BDIM_B * NDIM * 384;
  if (i >= total) return;
  int b = (int)(i / (NDIM * 384));
  int r = (int)(i % (NDIM * 384));
  int n = r / 384, c = r % 384;
  float v = (c < TDIM) ? xn[((long long)b * NDIM + n) * TDIM + c]
                       : feat[(long long)b * 288 + (c - TDIM)];
  h0[i] = v;
}

__global__ void fill_kernel(float* p, long long n, float v) {
  long long i = (long long)blockIdx.x * blockDim.x + threadIdx.x;
  if (i < n) p[i] = v;
}

__global__ void gru_gate_kernel(const float* __restrict__ GI, const float* __restrict__ GH,
                                const float* __restrict__ xT, float* __restrict__ h,
                                float* __restrict__ hst, int t) {
  int i = blockIdx.x * blockDim.x + threadIdx.x;
  if (i >= BDIM_B * NDIM) return;
  int b = i / NDIM, j = i % NDIM;
  const float* gi = GI + ((long long)b * TDIM + t) * 1536;
  const float* gh = GH + (long long)b * 1536;
  float ir = gi[j], iz = gi[512 + j], inn = gi[1024 + j];
  float hr = gh[j], hz = gh[512 + j], hn = gh[1024 + j];
  float r = 1.f / (1.f + expf(-(ir + hr)));
  float z = 1.f / (1.f + expf(-(iz + hz)));
  float ng = tanhf(inn + r * hn);
  float hp = h[i];
  float hnew = (1.f - z) * ng + z * hp;
  h[i] = hnew;
  long long o = ((long long)b * NDIM + j) * TDIM + t;
  hst[o] = hnew + xT[o];       // skip connection folded in
}

// fused time_dec resblock: rows = B*N*H, in = [fut(2), dec(8)], 10 -> 512 -> 1 (+skip 10->1)
__global__ __launch_bounds__(256)
void timedec_kernel(const float* __restrict__ decb, const float* __restrict__ feat,
                    const float* __restrict__ w1, const float* __restrict__ b1,
                    const float* __restrict__ w2, const float* __restrict__ b2,
                    const float* __restrict__ wsk, const float* __restrict__ bsk,
                    float* __restrict__ td) {
  __shared__ float sw1[10 * 512];
  __shared__ float sb1[512];
  __shared__ float sw2[512];
  int tid = threadIdx.x;
  for (int i = tid; i < 10 * 512; i += 256) sw1[i] = w1[i];
  for (int i = tid; i < 512; i += 256) { sb1[i] = b1[i]; sw2[i] = w2[i]; }
  __syncthreads();
  long long wid = ((long long)blockIdx.x * 256 + tid) >> 5;
  int lane = tid & 31;
  const long long total = (long long)BDIM_B * NDIM * HDIM;
  if (wid >= total) return;
  int b = (int)(wid / (NDIM * HDIM));
  int r = (int)(wid % (NDIM * HDIM));
  int n = r / HDIM, hh = r % HDIM;
  float in[10];
  in[0] = feat[((long long)b * 144 + TDIM + hh) * 2 + 0];
  in[1] = feat[((long long)b * 144 + TDIM + hh) * 2 + 1];
  const float* dp = decb + ((long long)b * NDIM + n) * 384 + hh * 8;
  for (int c = 0; c < 8; ++c) in[2 + c] = dp[c];
  float acc = 0.f;
  for (int j = lane; j < 512; j += 32) {
    float hsum = sb1[j];
#pragma unroll
    for (int k = 0; k < 10; ++k) hsum += in[k] * sw1[k * 512 + j];
    acc += (hsum > 0.f ? hsum : 0.f) * sw2[j];
  }
  for (int off = 16; off; off >>= 1) acc += __shfl_xor(acc, off, 32);
  if (lane == 0) {
    float sk = bsk[0];
    for (int k = 0; k < 10; ++k) sk += in[k] * wsk[k];
    td[wid] = acc + b2[0] + sk;
  }
}

__global__ void combine_kernel(const float* __restrict__ Cf, const float* __restrict__ fs,
                               const float* __restrict__ srcr, const float* __restrict__ meanb,
                               const float* __restrict__ stdb, float* __restrict__ out) {
  long long i = (long long)blockIdx.x * blockDim.x + threadIdx.x;
  const long long total = (long long)BDIM_B * NDIM * HDIM;
  if (i >= total) return;
  long long bn = i / HDIM;
  out[i] = Cf[i] + fs[i] + srcr[i] * stdb[bn] + meanb[bn];
}

// ---------------------------------------------------------------------------

extern "C" void kernel_launch(void* const* d_in, const int* in_sizes, int n_in,
                              void* d_out, int out_size, void* d_ws, size_t ws_size,
                              hipStream_t stream) {
  (void)in_sizes; (void)n_in; (void)out_size; (void)ws_size;
  auto F = [&](int i) { return (const float*)d_in[i]; };

  // pytree-flatten order (sorted dict keys at each level):
  // 0 batch_y_mark, 1 mul_L,
  // dec0: 2..7 (fc1.b,fc1.w,fc2.b,fc2.w,skip.b,skip.w), dec1: 8..13,
  // enc0: 14..19, enc1: 20..25, feat_enc: 26..31,
  // forecast: 32 b, 33 w, 34 gconv,
  // gru: 35 bhh, 36 bih, 37 lin.b, 38 lin.w, 39 whh, 40 wih,
  // 41 residual_proj.b, 42 residual_proj.w, 43 sepc_ln.b, 44 sepc_ln.w, 45 spec_weight,
  // tcn blocks j: base 46+6j = c1.b,c1.w,c2.b,c2.w,down.b,down.w; tcn.out: 70 b, 71 w,
  // time_dec: 72..77, 78 x
  const float* ymark = F(0);
  const float* mulL  = F(1);
  const float* gconv = F(34);
  const float* specw = F(45);
  const float* x     = F(78);

  char* base = (char*)d_ws;
  size_t off = 0;
  auto alloc = [&](long long elems) -> float* {
    float* p = (float*)(base + off);
    off += ((size_t)elems * sizeof(float) + 255) & ~(size_t)255;
    return p;
  };

  const long long BN = (long long)BDIM_B * NDIM;   // 16384 rows

  float* Cr     = alloc(96 * 96);
  float* Ci     = alloc(96 * 96);
  float* xT     = alloc(BN * 96);
  float* realb  = alloc(BN * 96);
  float* imagb  = alloc(BN * 96);
  float* T1     = alloc(BN * 512);
  float* T2     = alloc(BN * 512);
  float* T3     = alloc(BN * 512);
  float* specpre= alloc(BN * 96);
  float* specb  = alloc(BN * 96);
  float* ispecb = alloc(BN * 96);
  float* Mmat   = alloc(512 * 512);
  float* g1     = alloc(BN * 96);
  float* g2     = alloc(BN * 96);
  float* GI     = alloc(3072LL * 1536);
  float* GH     = alloc(32 * 1536);
  float* hcur   = alloc(32 * 512);
  float* hst    = alloc(BN * 96);
  float* fsb    = alloc(BN * 48);
  float* featH  = alloc(4608LL * 512);
  float* feat2  = alloc(4608 * 2);
  float* featb  = alloc(4608 * 2);
  float* meanb  = alloc(BN);
  float* stdb   = alloc(BN);
  float* xn     = alloc(BN * 96);
  float* tdb    = alloc(BN * 48);
  float* srcb   = alloc(BN * 48);
  float* Cfb    = alloc(BN * 48);

  // ---- spectral path: DFT (as GEMM) -> TCN -> iDFT -> sepc_ln -> ispec ----
  gen_dft_kernel<<<(96 * 96 + 255) / 256, 256, 0, stream>>>(Cr, Ci);
  transpose_btn_kernel<<<(int)((BN * 96 + 255) / 256), 256, 0, stream>>>(x, xT);
  gemm(stream, xT, 0, 96, 1, Cr, 0, 96, 1, nullptr, nullptr, 0, 0, realb, 0, 96,
       (int)BN, 96, 96, 1, 0);
  gemm(stream, xT, 0, 96, 1, Ci, 0, 96, 1, nullptr, nullptr, 0, 0, imagb, 0, 96,
       (int)BN, 96, 96, 1, 0);

  const int ch[5] = {96, 64, 128, 256, 512};
  auto run_tcn = [&](float* inout) {
    const float* h = inout;
    for (int j = 0; j < 4; ++j) {
      int Cin = ch[j], Co = ch[j + 1], d = 1 << j;
      int bi = 46 + j * 6;
      float* cand[3] = {T1, T2, T3};
      float* pick[2]; int pc = 0;
      for (int q = 0; q < 3; ++q) if (cand[q] != (float*)h && pc < 2) pick[pc++] = cand[q];
      float* y1 = pick[0]; float* y2 = pick[1];
      for (int kk = 0; kk < 3; ++kk) {                 // c1: 3 causal taps
        int last = (kk == 2);
        gemm(stream, h, 512LL * Cin, Cin, 1,
             F(bi + 1) + kk, 0, 3, 3LL * Cin,
             last ? F(bi) : nullptr,
             kk ? y1 : nullptr, 512LL * Co, Co,
             y1, 512LL * Co, Co, 512, Co, Cin, 32, last ? 1 : 0,
             -(2 - kk) * d, 0, 512);
      }
      for (int kk = 0; kk < 3; ++kk) {                 // c2
        int last = (kk == 2);
        gemm(stream, y1, 512LL * Co, Co, 1,
             F(bi + 3) + kk, 0, 3, 3LL * Co,
             last ? F(bi + 2) : nullptr,
             kk ? y2 : nullptr, 512LL * Co, Co,
             y2, 512LL * Co, Co, 512, Co, Co, 32, last ? 1 : 0,
             -(2 - kk) * d, 0, 512);
      }
      // h' = relu(down(h) + y2)
      gemm(stream, h, 512LL * Cin, Cin, 1,
           F(bi + 5), 0, 1, Cin,
           F(bi + 4),
           y2, 512LL * Co, Co,
           y1, 512LL * Co, Co, 512, Co, Cin, 32, 1, 0, 0, 512);
      h = y1;
    }
    gemm(stream, h, 0, 512, 1, F(71), 0, 96, 1, F(70), nullptr, 0, 0,
         inout, 0, 96, (int)BN, 96, 512, 1, 0);       // tcn.out 512->96
  };
  run_tcn(realb);
  run_tcn(imagb);

  ifft24_kernel<<<(int)(((long long)BDIM_B * 4 * NDIM * 24 + 255) / 256), 256, 0, stream>>>(
      realb, imagb, specpre);
  gemm(stream, specpre, 0, 96, 1, F(44), 0, 96, 1, F(43), nullptr, 0, 0,
       specb, 0, 96, (int)BN, 96, 96, 1, 0);
  ispec_kernel<<<(int)((BN + 255) / 256), 256, 0, stream>>>(specb, specw, ispecb);

  // ---- graph convolution: Mmat = sum_k gconv_k^T @ mul_L_k ; xg = M^4 @ xT ----
  for (int k = 0; k < 4; ++k) {
    gemm(stream, gconv + (long long)k * 512 * 512, 0, 1, 512,
         mulL + (long long)k * 512 * 512, 0, 512, 1,
         nullptr, (k == 0) ? nullptr : Mmat, 0, 512,
         Mmat, 0, 512, 512, 512, 512, 1, 0);
  }
  const float* xg = xT;
  float* nxt = g1;
  for (int it = 0; it < 4; ++it) {
    gemm(stream, Mmat, 0, 512, 1, xg, 512LL * 96, 96, 1, nullptr, nullptr, 0, 0,
         nxt, 512LL * 96, 96, 512, 96, 512, 32, 0);
    xg = nxt;
    nxt = (nxt == g1) ? g2 : g1;
  }

  // ---- skip-GRU: GI hoisted to one GEMM; 96 sequential h@Whh steps ----
  gemm(stream, x, 0, 512, 1, F(40), 0, 1536, 1, F(36), nullptr, 0, 0,
       GI, 0, 1536, 3072, 1536, 512, 1, 0);
  fill_kernel<<<(32 * 512 + 255) / 256, 256, 0, stream>>>(hcur, 32 * 512, 0.f);
  for (int t = 0; t < TDIM; ++t) {
    gemm(stream, hcur, 0, 512, 1, F(39), 0, 1536, 1, F(35), nullptr, 0, 0,
         GH, 0, 1536, 32, 1536, 512, 1, 0);
    gru_gate_kernel<<<(32 * 512 + 255) / 256, 256, 0, stream>>>(GI, GH, xT, hcur, hst, t);
  }
  gemm(stream, hst, 0, 96, 1, F(38), 0, 48, 1, F(37), nullptr, 0, 0,
       fsb, 0, 48, (int)BN, 48, 96, 1, 0);

  // ---- residual_connect ----
  meanvar_kernel<<<(int)((BN + 255) / 256), 256, 0, stream>>>(xT, xn, meanb, stdb);

  auto resblock = [&](const float* in, int M, int Kin, int Hd, int Kout, int ib,
                      float* th, float* tk, float* out) {
    gemm(stream, in, 0, Kin, 1, F(ib + 1), 0, Hd, 1, F(ib), nullptr, 0, 0,
         th, 0, Hd, M, Hd, Kin, 1, 1);
    gemm(stream, th, 0, Hd, 1, F(ib + 3), 0, Kout, 1, F(ib + 2), nullptr, 0, 0,
         tk, 0, Kout, M, Kout, Hd, 1, 0);
    gemm(stream, in, 0, Kin, 1, F(ib + 5), 0, Kout, 1, F(ib + 4), tk, 0, Kout,
         out, 0, Kout, M, Kout, Kin, 1, 0);
  };

  resblock(ymark, 4608, 4, 512, 2, 26, featH, feat2, featb);              // feat_enc
  concat_enc_kernel<<<(int)((BN * 384 + 255) / 256), 256, 0, stream>>>(xn, featb, T1);
  resblock(T1, (int)BN, 384, 512, 512, 14, T2, T3, T2);                   // enc0
  resblock(T2, (int)BN, 512, 512, 512, 20, T3, T1, T3);                   // enc1
  resblock(T3, (int)BN, 512, 512, 512, 2,  T1, T2, T1);                   // dec0
  resblock(T1, (int)BN, 512, 512, 384, 8,  T2, T3, T2);                   // dec1 -> decbuf=T2

  {
    long long rows = BN * HDIM;                    // one wave per output scalar
    long long blocks = (rows * 32 + 255) / 256;
    timedec_kernel<<<(int)blocks, 256, 0, stream>>>(T2, featb, F(73), F(72), F(75),
                                                    F(74), F(77), F(76), tdb);
  }
  // src_raw = xn @ residual_proj + b + td   (scale/shift folded into combine)
  gemm(stream, xn, 0, 96, 1, F(42), 0, 48, 1, F(41), tdb, 0, 48,
       srcb, 0, 48, (int)BN, 48, 96, 1, 0);

  // ---- forecast head: split-K over concat([ispec, xg]) ----
  gemm(stream, ispecb, 0, 96, 1, F(33), 0, 48, 1, nullptr, nullptr, 0, 0,
       Cfb, 0, 48, (int)BN, 48, 96, 1, 0);
  gemm(stream, xg, 0, 96, 1, F(33) + 96 * 48, 0, 48, 1, F(32), Cfb, 0, 48,
       Cfb, 0, 48, (int)BN, 48, 96, 1, 0);

  combine_kernel<<<(int)((BN * HDIM + 255) / 256), 256, 0, stream>>>(
      Cfb, fsb, srcb, meanb, stdb, (float*)d_out);
}
